// SparseConvLSTMCell_26285199851964
// MI455X (gfx1250) — matrix operand order repaired and checked by reading, not compiled, not run on hardware
//
#include <hip/hip_runtime.h>
#include <hip/hip_bf16.h>

// ---------------------------------------------------------------------------
// Problem constants (from the reference)
// ---------------------------------------------------------------------------
#define NPTS   524288
#define BDIM   2
#define DDIM   64
#define HDIM   256
#define WDIM   256
#define MAPSZ  (BDIM * DDIM * HDIM * WDIM)   // 8,388,608
#define C_IN   64
#define C_OUT  128
#define NOFF   27
#define PW_PER_K 8192                        // bf16 elems per offset: 2*8*512

// Workspace layout (bytes)
#define WS_MAP_OFF    0u
#define WS_FEAT_OFF   33554432u                 // MAPSZ * 4
#define WS_PW_OFF     (33554432u + 67108864u)   // + NPTS*64*2

// ---------------------------------------------------------------------------
// CDNA5 WMMA types
// ---------------------------------------------------------------------------
typedef __attribute__((ext_vector_type(16))) __bf16        v16bf;
typedef __attribute__((ext_vector_type(8)))  float         v8f;
typedef __attribute__((ext_vector_type(4)))  unsigned int  v4u;

union FragU {
    v4u   q[2];   // two 16B halves
    v16bf v;      // bf16 WMMA operand (8 VGPRs)
};

// float -> bf16 (RNE), bit-level
__device__ __forceinline__ unsigned short f2bf(float f) {
    unsigned int u = __builtin_bit_cast(unsigned int, f);
    unsigned int r = u + 0x7fffu + ((u >> 16) & 1u);
    return (unsigned short)(r >> 16);
}

__device__ __forceinline__ float fast_sigmoid(float x) {
    return __builtin_amdgcn_rcpf(1.0f + __expf(-x));
}
__device__ __forceinline__ float fast_tanh(float x) {
    return 2.0f * __builtin_amdgcn_rcpf(1.0f + __expf(-2.0f * x)) - 1.0f;
}

// ---------------------------------------------------------------------------
// CDNA5 async global->LDS copy (ASYNCcnt-tracked) + wait, via inline asm so it
// works on both ROCm 7.2 and amdgpu-toolchain regardless of builtin arity.
// LDS byte address = low 32 bits of the generic pointer to a __shared__ object.
// ---------------------------------------------------------------------------
__device__ __forceinline__ void async_copy_b128(const unsigned short* g,
                                                const unsigned short* l) {
    unsigned lds_off = (unsigned)(unsigned long long)(const void*)l;
    unsigned long long gaddr = (unsigned long long)(const void*)g;
    asm volatile("global_load_async_to_lds_b128 %0, %1, off"
                 :: "v"(lds_off), "v"(gaddr) : "memory");
}
__device__ __forceinline__ void wait_async0() {
    asm volatile("s_wait_asynccnt 0x0" ::: "memory");
}

// ---------------------------------------------------------------------------
// Kernel 1: idx_map = -1
// ---------------------------------------------------------------------------
__global__ __launch_bounds__(256) void init_map_kernel(int* __restrict__ idx_map) {
    int i = blockIdx.x * 256 + threadIdx.x;   // grid sized exactly to MAPSZ
    idx_map[i] = -1;
}

// ---------------------------------------------------------------------------
// Kernel 2: scatter point ids into the dense coordinate map
// ---------------------------------------------------------------------------
__global__ __launch_bounds__(256) void scatter_kernel(const int* __restrict__ coors,
                                                      int* __restrict__ idx_map) {
    int i = blockIdx.x * 256 + threadIdx.x;   // grid == NPTS
    int b = coors[i * 4 + 0];
    int z = coors[i * 4 + 1];
    int y = coors[i * 4 + 2];
    int x = coors[i * 4 + 3];
    idx_map[((b * DDIM + z) * HDIM + y) * WDIM + x] = i;
}

// ---------------------------------------------------------------------------
// Kernel 3: pack [x || h] -> bf16 rows of 64
// ---------------------------------------------------------------------------
__global__ __launch_bounds__(256) void pack_feats_kernel(const float* __restrict__ xf,
                                                         const float* __restrict__ hf,
                                                         unsigned short* __restrict__ featsBF) {
    int idx = blockIdx.x * 256 + threadIdx.x;   // grid == NPTS*64
    int i = idx >> 6;
    int j = idx & 63;
    float v = (j < 32) ? xf[(i << 5) + j] : hf[(i << 5) + (j - 32)];
    featsBF[idx] = f2bf(v);
}

// ---------------------------------------------------------------------------
// Kernel 4: pack weight[27][64][128] f32 -> bf16 fragment-native B layout.
// Per fragment (k,c,t): 1024 bytes =
//   [  0..511]: per-lane lo halves  (lane*16B) = B components e=0..7
//   [512..1023]: per-lane hi halves (lane*16B) = B components e=8..15
// 16B lane stride minimizes LDS bank-conflict multiplicity for ds_load_b128.
// B element e -> K = c*32 + (e>>3)*16 + (lane>>4)*8 + (e&7),  N = t*16 + (lane&15).
// ---------------------------------------------------------------------------
__global__ __launch_bounds__(256) void pack_weight_kernel(const float* __restrict__ wt,
                                                          unsigned short* __restrict__ pw) {
    int idx = blockIdx.x * 256 + threadIdx.x;   // grid == 27*2*8*512 = 221184
    int w    =  idx        & 511;               // offset within fragment
    int t    = (idx >> 9)  & 7;
    int c    = (idx >> 12) & 1;
    int k    =  idx >> 13;
    int hiR  =  w >> 8;                         // 0 = lo region, 1 = hi region
    int lane = (w >> 3) & 31;
    int e    =  hiR * 8 + (w & 7);
    int Kg   = c * 32 + hiR * 16 + (lane >> 4) * 8 + (w & 7);   // 0..63
    int Nc   = t * 16 + (lane & 15);            // 0..127
    pw[idx] = f2bf(wt[(k * C_IN + Kg) * C_OUT + Nc]);
}

// ---------------------------------------------------------------------------
// Neighbor lookup + A-fragment gather helpers
// ---------------------------------------------------------------------------
__device__ __forceinline__ int neighbor_id(int k, int cb, int cz, int cy, int cx,
                                           const int* __restrict__ idx_map) {
    int dz = k / 9 - 1;
    int dy = (k / 3) % 3 - 1;
    int dx = k % 3 - 1;
    int nz = cz + dz, ny = cy + dy, nx = cx + dx;
    int nid = -1;
    if ((unsigned)nz < (unsigned)DDIM && (unsigned)ny < (unsigned)HDIM &&
        (unsigned)nx < (unsigned)WDIM) {
        nid = idx_map[((cb * DDIM + nz) * HDIM + ny) * WDIM + nx];
    }
    return nid;
}

__device__ __forceinline__ void gatherA(v4u a[4], const unsigned short* __restrict__ feats,
                                        int nid, int kbaseA) {
    const unsigned short* ap = feats + ((long)(nid < 0 ? 0 : nid) << 6) + kbaseA;
    a[0] = *(const v4u*)(ap);        // chunk0, K = kbase..kbase+7
    a[1] = *(const v4u*)(ap + 16);   // chunk0, K = 16+kbase..
    a[2] = *(const v4u*)(ap + 32);   // chunk1, K = 32+kbase..
    a[3] = *(const v4u*)(ap + 48);   // chunk1, K = 48+kbase..
}

// ---------------------------------------------------------------------------
// Kernel 5: gather -> WMMA bf16 GEMM over 27 offsets -> fused LSTM gating.
// One wave32 per 16 points. Weights double-buffered in LDS via async copies;
// sparse A gathers software-pipelined one offset ahead.
// ---------------------------------------------------------------------------
__global__ __launch_bounds__(256) void subm_conv_lstm_wmma(
    const unsigned short* __restrict__ featsBF,
    const unsigned short* __restrict__ pw,
    const int*            __restrict__ idx_map,
    const int*            __restrict__ coors,
    const float*          __restrict__ bias,
    const float*          __restrict__ c_prev,
    float*                __restrict__ outp) {

    __shared__ unsigned short smem[2][PW_PER_K];   // 2 x 16 KiB weight stage

    const int tid   = threadIdx.x;
    const int lane  = tid & 31;
    const int wave  = tid >> 5;
    const int gw    = blockIdx.x * 8 + wave;       // 4096 blocks * 8 waves
    const int row0  = gw << 4;                     // 16 points per wave
    const int sub   = lane & 15;
    const int half  = lane >> 4;
    const int myrow = row0 + sub;
    const int kbaseA = half * 8;

    // coordinates of this lane's point (lanes 16-31 mirror 0-15)
    const int cb = coors[myrow * 4 + 0];
    const int cz = coors[myrow * 4 + 1];
    const int cy = coors[myrow * 4 + 2];
    const int cx = coors[myrow * 4 + 3];

    // accumulators init with bias: tile t = output columns [t*16, t*16+16)
    v8f acc[8];
#pragma unroll
    for (int t = 0; t < 8; ++t) {
        float bv = bias[t * 16 + sub];
#pragma unroll
        for (int j = 0; j < 8; ++j) acc[t][j] = bv;
    }

    const v4u zz = {0u, 0u, 0u, 0u};

    // ---- prologue: stage weights for k=0, gather A for k=0 ----
#pragma unroll
    for (int i = 0; i < 4; ++i) {
        int off = (tid + i * 256) * 8;             // 16B chunks
        async_copy_b128(pw + off, &smem[0][off]);
    }
    int  nid0 = neighbor_id(0, cb, cz, cy, cx, idx_map);
    bool vcur = (nid0 >= 0);
    v4u  acur[4];
    gatherA(acur, featsBF, nid0, kbaseA);

#pragma unroll 1
    for (int k = 0; k < NOFF; ++k) {
        // current LDS buffer complete (our async loads done) + all waves ready
        wait_async0();
        __syncthreads();

        // kick off next weight stage (buffer last read before the barrier above)
        if (k + 1 < NOFF) {
            const unsigned short* pwk = pw + (k + 1) * PW_PER_K;
            unsigned short*       dst = smem[(k + 1) & 1];
#pragma unroll
            for (int i = 0; i < 4; ++i) {
                int off = (tid + i * 256) * 8;
                async_copy_b128(pwk + off, dst + off);
            }
        }

        // kick off next sparse A gather (overlaps with the WMMA burst below)
        v4u  anext[4] = {zz, zz, zz, zz};
        bool vnext = false;
        if (k + 1 < NOFF) {
            int nid = neighbor_id(k + 1, cb, cz, cy, cx, idx_map);
            vnext = (nid >= 0);
            gatherA(anext, featsBF, nid, kbaseA);
        }

        // compute with current A + staged B
        const unsigned short* bufc = smem[k & 1];
#pragma unroll
        for (int c = 0; c < 2; ++c) {
            FragU au;
            au.q[0] = vcur ? acur[c * 2 + 0] : zz;
            au.q[1] = vcur ? acur[c * 2 + 1] : zz;
#pragma unroll
            for (int t = 0; t < 8; ++t) {
                const unsigned short* fb = bufc + (c * 8 + t) * 512;
                FragU bu;
                bu.q[0] = *(const v4u*)(fb + lane * 8);          // lo region
                bu.q[1] = *(const v4u*)(fb + 256 + lane * 8);    // hi region
                acc[t] = __builtin_amdgcn_wmma_f32_16x16x32_bf16(
                    false, au.v, false, bu.v, (short)0, acc[t], false, false);
            }
        }

        // rotate pipeline
#pragma unroll
        for (int i = 0; i < 4; ++i) acur[i] = anext[i];
        vcur = vnext;
    }

    // ---- fused LSTM gating straight from accumulator layout ----
    // acc tile t, VGPR j holds (row = j + 8*half, col = t*16 + sub).
#pragma unroll
    for (int t = 0; t < 2; ++t) {
#pragma unroll
        for (int j = 0; j < 8; ++j) {
            const int m   = row0 + half * 8 + j;
            const int cix = t * 16 + sub;                 // 0..31
            float iv = fast_sigmoid(acc[t    ][j]);
            float fv = fast_sigmoid(acc[t + 2][j]);
            float ov = fast_sigmoid(acc[t + 4][j]);
            float gv = fast_tanh   (acc[t + 6][j]);
            float cp = c_prev[m * 32 + cix];
            float cn = fv * cp + iv * gv;
            float hn = ov * fast_tanh(cn);
            outp[m * 32 + cix]             = hn;          // h_next
            outp[NPTS * 32 + m * 32 + cix] = cn;          // c_next
        }
    }
}

// ---------------------------------------------------------------------------
// Host-side launcher (graph-capture safe: kernels only, all on `stream`)
// ---------------------------------------------------------------------------
extern "C" void kernel_launch(void* const* d_in, const int* in_sizes, int n_in,
                              void* d_out, int out_size, void* d_ws, size_t ws_size,
                              hipStream_t stream) {
    (void)in_sizes; (void)n_in; (void)out_size; (void)ws_size;

    const float* xf  = (const float*)d_in[0];   // (N, 32)
    const float* hf  = (const float*)d_in[1];   // (N, 32)
    const float* cf  = (const float*)d_in[2];   // (N, 32)
    const float* wt  = (const float*)d_in[3];   // (27, 64, 128)
    const float* bs  = (const float*)d_in[4];   // (128,)
    const int*   coo = (const int*)d_in[5];     // (N, 4)
    float*       out = (float*)d_out;           // h_next (N*32) || c_next (N*32)

    char* ws = (char*)d_ws;
    int*            idx_map = (int*)(ws + WS_MAP_OFF);              // 32 MiB
    unsigned short* featsBF = (unsigned short*)(ws + WS_FEAT_OFF);  // 64 MiB
    unsigned short* pw      = (unsigned short*)(ws + WS_PW_OFF);    // 432 KiB

    init_map_kernel   <<<MAPSZ / 256,        256, 0, stream>>>(idx_map);
    scatter_kernel    <<<NPTS / 256,         256, 0, stream>>>(coo, idx_map);
    pack_feats_kernel <<<(NPTS * 64) / 256,  256, 0, stream>>>(xf, hf, featsBF);
    pack_weight_kernel<<<221184 / 256,       256, 0, stream>>>(wt, pw);
    subm_conv_lstm_wmma<<<NPTS / (16 * 8),   256, 0, stream>>>(featsBF, pw, idx_map,
                                                               coo, bs, cf, out);
}
